// HierarchicalAttention_85263690760730
// MI455X (gfx1250) — matrix operand order repaired and checked by reading, not compiled, not run on hardware
//
#include <hip/hip_runtime.h>

// ---------------------------------------------------------------------------
// Hierarchical Bahdanau attention for MI455X (gfx1250), bf16 WMMA path.
// B=32, T=4096, H=512, CHUNK=32, n_chunks=128.
// v2: A tile staged in LDS as bf16 once (no cvt in the WMMA loop),
//     shfl_xor score reduction, global_prefetch on the B stream.
// ---------------------------------------------------------------------------

typedef __attribute__((ext_vector_type(16))) __bf16 v16bf;
typedef __attribute__((ext_vector_type(8)))  float  v8f;

#define HDIM   512
#define LDST   516      // padded f32 LDS row stride (16B aligned, bank-skewed)
#define BSTR   520      // padded bf16 LDS row stride (16B aligned, bank-skewed)

// ---------------- WMMA helpers --------------------------------------------

__device__ __forceinline__ v8f wmma_bf16(v16bf a, v16bf b, v8f c) {
    return __builtin_amdgcn_wmma_f32_16x16x32_bf16(
        /*neg_a=*/false, a, /*neg_b=*/false, b,
        /*c_mod=*/(short)0, c, /*reuse_a=*/false, /*reuse_b=*/false);
}

__device__ __forceinline__ v16bf load16bf(const __bf16* __restrict__ p) {
    union { uint4 q[2]; v16bf v; } u;
    const uint4* p4 = (const uint4*)p;
    u.q[0] = p4[0];
    u.q[1] = p4[1];
    return u.v;
}

// A fragment: 16-bit A-matrix 16x32, from bf16 LDS tile (contiguous K order, matches B).
// lanes 0-15  (M = lane):    elems 0-15 = K kk+0..15
// lanes 16-31 (M = lane-16): elems 0-15 = K kk+16..31
__device__ __forceinline__ v16bf load_a_frag(const __bf16* __restrict__ xbf,
                                             int mtile, int kk, int lane) {
    const int row = mtile * 16 + (lane & 15);
    const int kb  = kk + ((lane >> 4) << 4);
    return load16bf(xbf + row * BSTR + kb);
}

// B fragment pointer into pre-swizzled weights.
// Layout: per k-block (k/32) and column n, 64 contiguous bytes of B[k..k+31][n].
__device__ __forceinline__ const __bf16* bptr(const __bf16* __restrict__ Bsw,
                                              int col, int kk, int lane) {
    return Bsw + (((size_t)(kk >> 5) * (size_t)HDIM + (size_t)col) << 5)
               + ((lane >> 4) << 4);
}

// Core tile engine: one wave computes a 32(M) x 64(N) x 512(K) GEMM slab.
__device__ __forceinline__ void gemm32x64(const __bf16* __restrict__ xbf,
                                          const __bf16* __restrict__ Bsw,
                                          int c0, int lane, v8f (&acc)[2][4]) {
    #pragma unroll
    for (int m = 0; m < 2; ++m)
        #pragma unroll
        for (int n = 0; n < 4; ++n)
            #pragma unroll
            for (int r = 0; r < 8; ++r)
                acc[m][n][r] = 0.0f;

    for (int kk = 0; kk < HDIM; kk += 32) {
        v16bf a0 = load_a_frag(xbf, 0, kk, lane);
        v16bf a1 = load_a_frag(xbf, 1, kk, lane);
        #pragma unroll
        for (int n = 0; n < 4; ++n) {
            const __bf16* p = bptr(Bsw, c0 + n * 16 + (lane & 15), kk, lane);
            if (n == 0)  // stream 2 K-blocks (64 KB) ahead -> global_prefetch
                __builtin_prefetch((const void*)(p + 32768), 0, 1);
            v16bf b = load16bf(p);
            acc[0][n] = wmma_bf16(a0, b, acc[0][n]);
            acc[1][n] = wmma_bf16(a1, b, acc[1][n]);
        }
    }
}

// Stage 32x512 f32 rows (stride rowStride) into bf16 LDS tile.
__device__ __forceinline__ void stage_a_bf16(__bf16* __restrict__ xbf,
                                             const float* __restrict__ rowbase,
                                             long rowStride, int row0, int tid) {
    for (int i = tid; i < 32 * (HDIM / 2); i += 256) {
        const int t = i >> 8;
        const int j = (i & 255) << 1;
        float2 f = *(const float2*)(rowbase + (size_t)(row0 + t) * (size_t)rowStride + j);
        union { __bf16 h[2]; unsigned u; } pk;
        pk.h[0] = (__bf16)f.x;
        pk.h[1] = (__bf16)f.y;
        *(unsigned*)&xbf[t * BSTR + j] = pk.u;
    }
}

// energy -> per-lane score partials -> butterfly reduce -> LDS atomics
__device__ __forceinline__ void score_reduce(const v8f (&acc)[2][4],
                                             const float* __restrict__ qv,
                                             const float* __restrict__ vv,
                                             float* __restrict__ sc,
                                             int c0, int lane) {
    const int moff = (lane >> 4) << 3;
    float pt[2][8];
    #pragma unroll
    for (int m = 0; m < 2; ++m)
        #pragma unroll
        for (int r = 0; r < 8; ++r) pt[m][r] = 0.0f;

    #pragma unroll
    for (int m = 0; m < 2; ++m)
        #pragma unroll
        for (int n = 0; n < 4; ++n) {
            const int h = c0 + n * 16 + (lane & 15);
            const float qh = qv[h], vh = vv[h];
            #pragma unroll
            for (int r = 0; r < 8; ++r)
                pt[m][r] += tanhf(qh + acc[m][n][r]) * vh;
        }

    // reduce over the 16 lanes of each half-wave (they hold the same rows,
    // different columns)
    #pragma unroll
    for (int m = 0; m < 2; ++m)
        #pragma unroll
        for (int r = 0; r < 8; ++r) {
            float s = pt[m][r];
            s += __shfl_xor(s, 1, 16);
            s += __shfl_xor(s, 2, 16);
            s += __shfl_xor(s, 4, 16);
            s += __shfl_xor(s, 8, 16);
            pt[m][r] = s;
        }
    if ((lane & 15) == 0) {
        #pragma unroll
        for (int m = 0; m < 2; ++m)
            #pragma unroll
            for (int r = 0; r < 8; ++r)
                atomicAdd(&sc[m * 16 + moff + r], pt[m][r]);
    }
}

// ---------------- K1: weight swizzle f32 -> bf16 fragment layout ----------

__global__ void conv_bsw(const float* __restrict__ s0, const float* __restrict__ s1,
                         const float* __restrict__ s2, const float* __restrict__ s3,
                         __bf16* __restrict__ d0, __bf16* __restrict__ d1,
                         __bf16* __restrict__ d2, __bf16* __restrict__ d3) {
    const int which = blockIdx.y;
    const float* s = (which == 0) ? s0 : (which == 1) ? s1 : (which == 2) ? s2 : s3;
    __bf16*      d = (which == 0) ? d0 : (which == 1) ? d1 : (which == 2) ? d2 : d3;
    const int i = blockIdx.x * 256 + threadIdx.x;   // 512*512 elems
    const int j = i & (HDIM - 1);                   // K index
    const int h = i >> 9;                           // N index
    d[((size_t)(j >> 5) * HDIM + (size_t)h) * 32 + (j & 31)] =
        (__bf16)s[(size_t)h * HDIM + j];
}

// ---------------- batched query projection (q + b1 + b2) ------------------

__global__ void qgemm(const float* __restrict__ rowbase, long rowStride,
                      const __bf16* __restrict__ Bsw,
                      const float* __restrict__ b1, const float* __restrict__ b2,
                      float* __restrict__ out) {
    __shared__ __bf16 xbf[32 * BSTR];
    const int tid  = threadIdx.x;
    const int lane = tid & 31;
    const int c0   = (tid >> 5) * 64;
    const int row0 = blockIdx.x * 32;

    stage_a_bf16(xbf, rowbase, rowStride, row0, tid);
    __syncthreads();

    v8f acc[2][4];
    gemm32x64(xbf, Bsw, c0, lane, acc);

    const int moff = (lane >> 4) << 3;
    #pragma unroll
    for (int m = 0; m < 2; ++m)
        #pragma unroll
        for (int n = 0; n < 4; ++n) {
            const int h = c0 + n * 16 + (lane & 15);
            const float bias = b1[h] + b2[h];
            #pragma unroll
            for (int r = 0; r < 8; ++r) {
                const int t = m * 16 + moff + r;
                out[(size_t)(row0 + t) * HDIM + h] = acc[m][n][r] + bias;
            }
        }
}

// ---------------- stage-1 attention, one WG per (b,chunk) -----------------

__global__ void att32(const float* __restrict__ x, const __bf16* __restrict__ Usw,
                      const float* __restrict__ qc, const float* __restrict__ v,
                      float* __restrict__ sub_ctx) {
    __shared__ __bf16 xbf[32 * BSTR];
    __shared__ float  xf[32 * LDST];
    __shared__ float  qv[HDIM];
    __shared__ float  vv[HDIM];
    __shared__ float  sc[32];

    const int bc   = blockIdx.x;            // b*128 + chunk
    const int tid  = threadIdx.x;
    const int lane = tid & 31;
    const int c0   = (tid >> 5) * 64;

    const float* xg = x + (size_t)bc * 32 * HDIM;  // chunk rows are contiguous
    for (int i = tid; i < 32 * (HDIM / 2); i += 256) {
        const int t = i >> 8;
        const int j = (i & 255) << 1;
        float2 f = *(const float2*)(xg + (size_t)t * HDIM + j);
        xf[t * LDST + j]     = f.x;
        xf[t * LDST + j + 1] = f.y;
        union { __bf16 h[2]; unsigned u; } pk;
        pk.h[0] = (__bf16)f.x;
        pk.h[1] = (__bf16)f.y;
        *(unsigned*)&xbf[t * BSTR + j] = pk.u;
    }
    for (int i = tid; i < HDIM; i += 256) {
        qv[i] = qc[(size_t)bc * HDIM + i];
        vv[i] = v[i];
    }
    if (tid < 32) sc[tid] = 0.0f;
    __syncthreads();

    v8f acc[2][4];
    gemm32x64(xbf, Usw, c0, lane, acc);
    score_reduce(acc, qv, vv, sc, c0, lane);
    __syncthreads();

    // softmax weights (redundant per-thread; 32 elems is cheap)
    float mx = -3.0e38f;
    #pragma unroll
    for (int t = 0; t < 32; ++t) mx = fmaxf(mx, sc[t]);
    float w[32], sum = 0.0f;
    #pragma unroll
    for (int t = 0; t < 32; ++t) { w[t] = __expf(sc[t] - mx); sum += w[t]; }
    const float inv = 1.0f / sum;

    for (int h = tid; h < HDIM; h += 256) {
        float c = 0.0f;
        #pragma unroll
        for (int t = 0; t < 32; ++t) c += w[t] * xf[t * LDST + h];
        sub_ctx[(size_t)bc * HDIM + h] = c * inv;
    }
}

// ---------------- stage-2 scores, one WG per (b, 32-row block) ------------

__global__ void att_sc(const float* __restrict__ sub_ctx, const __bf16* __restrict__ Usw,
                       const float* __restrict__ qc, const float* __restrict__ v,
                       float* __restrict__ tscores) {
    __shared__ __bf16 xbf[32 * BSTR];
    __shared__ float  qv[HDIM];
    __shared__ float  vv[HDIM];
    __shared__ float  sc[32];

    const int b    = blockIdx.x >> 2;
    const int mb   = blockIdx.x & 3;
    const int tid  = threadIdx.x;
    const int lane = tid & 31;
    const int c0   = (tid >> 5) * 64;

    stage_a_bf16(xbf, sub_ctx + (size_t)(b * 128 + mb * 32) * HDIM,
                 (long)HDIM, 0, tid);
    for (int i = tid; i < HDIM; i += 256) {
        qv[i] = qc[(size_t)b * HDIM + i];
        vv[i] = v[i];
    }
    if (tid < 32) sc[tid] = 0.0f;
    __syncthreads();

    v8f acc[2][4];
    gemm32x64(xbf, Usw, c0, lane, acc);
    score_reduce(acc, qv, vv, sc, c0, lane);
    __syncthreads();

    if (tid < 32) tscores[b * 128 + mb * 32 + tid] = sc[tid];
}

// ---------------- stage-2 softmax + context -------------------------------

__global__ void finalize(const float* __restrict__ tscores,
                         const float* __restrict__ sub_ctx,
                         float* __restrict__ out) {
    __shared__ float al[128];
    const int b   = blockIdx.x;
    const int tid = threadIdx.x;

    if (tid < 128) al[tid] = tscores[b * 128 + tid];
    __syncthreads();

    float mx = -3.0e38f;
    for (int t = 0; t < 128; ++t) mx = fmaxf(mx, al[t]);
    float sum = 0.0f;
    for (int t = 0; t < 128; ++t) sum += __expf(al[t] - mx);
    const float inv = 1.0f / sum;

    float myw = 0.0f;
    if (tid < 128) myw = __expf(al[tid] - mx) * inv;
    __syncthreads();
    if (tid < 128) {
        al[tid] = myw;
        out[32 * HDIM + b * 128 + tid] = myw;    // top_alpha
    }
    __syncthreads();

    for (int h = tid; h < HDIM; h += 256) {
        const float* sp = sub_ctx + (size_t)b * 128 * HDIM + h;
        float c = 0.0f;
        for (int t = 0; t < 128; ++t) c += al[t] * sp[(size_t)t * HDIM];
        out[b * HDIM + h] = c;                   // top_ctx
    }
}

// ---------------- launcher ------------------------------------------------

extern "C" void kernel_launch(void* const* d_in, const int* in_sizes, int n_in,
                              void* d_out, int out_size, void* d_ws, size_t ws_size,
                              hipStream_t stream) {
    const float* lstm   = (const float*)d_in[0];
    const float* sub_W  = (const float*)d_in[1];
    const float* sub_Wb = (const float*)d_in[2];
    const float* sub_U  = (const float*)d_in[3];
    const float* sub_Ub = (const float*)d_in[4];
    const float* sub_v  = (const float*)d_in[5];
    // d_in[6] = sub_vb: constant wrt softmax -> unused
    const float* top_W  = (const float*)d_in[7];
    const float* top_Wb = (const float*)d_in[8];
    const float* top_U  = (const float*)d_in[9];
    const float* top_Ub = (const float*)d_in[10];
    const float* top_v  = (const float*)d_in[11];
    // d_in[12] = top_vb: unused (softmax invariant)

    // workspace carve-up (~18.1 MiB = ~18.96 MB total)
    char* ws = (char*)d_ws;
    __bf16* Usw  = (__bf16*)ws; ws += (size_t)HDIM * HDIM * 2;
    __bf16* Wsw  = (__bf16*)ws; ws += (size_t)HDIM * HDIM * 2;
    __bf16* TUsw = (__bf16*)ws; ws += (size_t)HDIM * HDIM * 2;
    __bf16* TWsw = (__bf16*)ws; ws += (size_t)HDIM * HDIM * 2;
    float* Qc      = (float*)ws; ws += (size_t)4096 * HDIM * 4;      // stage-1 q+Wb+Ub
    float* subctx  = (float*)ws; ws += (size_t)32 * 128 * HDIM * 4;  // (B,128,H)
    float* tqc     = (float*)ws; ws += (size_t)32 * HDIM * 4;        // stage-2 q
    float* tsc     = (float*)ws; ws += (size_t)32 * 128 * 4;         // stage-2 scores
    float* outp    = (float*)d_out;

    // 1) swizzle all four weight matrices into bf16 B-fragment layout
    conv_bsw<<<dim3((HDIM * HDIM) / 256, 4), 256, 0, stream>>>(
        sub_U, sub_W, top_U, top_W, Usw, Wsw, TUsw, TWsw);

    // 2) stage-1 queries: rows = x[b, c*32+31, :] for all 4096 (b,c)
    qgemm<<<128, 256, 0, stream>>>(lstm + 31 * HDIM, (long)(32 * HDIM),
                                   Wsw, sub_Wb, sub_Ub, Qc);

    // 3) stage-1 attention: one WG per (b,chunk)
    att32<<<4096, 256, 0, stream>>>(lstm, Usw, Qc, sub_v, subctx);

    // 4) stage-2 queries: rows = sub_ctx[b, 127, :] for b in 0..31
    qgemm<<<1, 256, 0, stream>>>(subctx + 127 * HDIM, (long)(128 * HDIM),
                                 TWsw, top_Wb, top_Ub, tqc);

    // 5) stage-2 scores
    att_sc<<<128, 256, 0, stream>>>(subctx, TUsw, tqc, top_v, tsc);

    // 6) softmax + top context
    finalize<<<32, 256, 0, stream>>>(tsc, subctx, outp);
}
